// GraphConvolution_1958505087040
// MI455X (gfx1250) — compile-verified, matrix operand order checked by reading, not run on hardware
//
#include <hip/hip_runtime.h>

// ---------------------------------------------------------------------------
// GCN layer for MI455X (gfx1250, wave32, WMMA).
//   out = D^-1/2 (A+I) D^-1/2 ((x*mask) @ W) + b
// Refactored:
//   dinv[i]   = rsqrt(#nonzeros in adj row i)
//   support   = (x*mask) @ W                        (f32, row-major)
//   s2t[c,j]  = f16( dinv[j] * support[j,c] )       (transposed, 2 MB, L2-hot)
//   out[i,:]  = dinv[i]*(adj_f16 @ s2)[i,:] + dinv[i]^2*support[i,:] + b
// Big GEMM (8192x8192x128): streams adj once (268 MB ~ 11.5us @ 23.3 TB/s);
// v_wmma_f32_16x16x32_f16 compute (~3.4us) stays under the HBM time, so the
// loop is software-pipelined (reg prefetch + double-buffered LDS + L2
// prefetch 8 tiles ahead) to actually sustain the bandwidth bound.
// ---------------------------------------------------------------------------

typedef _Float16 half_t;
typedef _Float16 v16h __attribute__((ext_vector_type(16)));
typedef _Float16 v8h  __attribute__((ext_vector_type(8)));
typedef float    v8f  __attribute__((ext_vector_type(8)));

#define NROW 8192
#define DIM  128
#define KT   32           // K tile (f32 K-row = 128 B = one cacheline)
#define LDA  40           // padded LDS stride (halves); 40*2=80 B, 16B aligned
#define PF_TILES 8        // L2 prefetch distance in K tiles

// Load one 16x32 f16 WMMA operand fragment from LDS (row-major, stride LDA).
// ISA 16-bit operand layout: lanes 0-15 hold K 0..7 (v0-3) and 16..23 (v4-7),
// lanes 16-31 hold K 8..15 and 24..31; caller passes base = row*LDA + kbase.
__device__ __forceinline__ v16h ld_frag(const half_t* p) {
    v8h lo = *(const v8h*)(p);
    v8h hi = *(const v8h*)(p + 16);
    v16h f;
#pragma unroll
    for (int i = 0; i < 8; ++i) { f[i] = lo[i]; f[i + 8] = hi[i]; }
    return f;
}

// ---------------- Pass 1: dinv[i] = rsqrt(sum(adj[i,:] > 0)) ----------------
__global__ __launch_bounds__(256) void gcn_degree(const float* __restrict__ adj,
                                                  float* __restrict__ dinv) {
    __shared__ float part[8];
    const int row = blockIdx.x;
    const int t   = threadIdx.x;
    const float4* rp = (const float4*)(adj + (size_t)row * NROW);
    float cnt = 0.0f;
#pragma unroll
    for (int i = 0; i < 8; ++i) {               // 8192 f32 = 2048 float4
        float4 v = rp[t + i * 256];             // coalesced
        cnt += (float)((v.x > 0.0f) + (v.y > 0.0f) + (v.z > 0.0f) + (v.w > 0.0f));
    }
#pragma unroll
    for (int off = 16; off > 0; off >>= 1) cnt += __shfl_down(cnt, off);
    if ((t & 31) == 0) part[t >> 5] = cnt;
    __syncthreads();
    if (t == 0) {
        float s = 0.0f;
#pragma unroll
        for (int i = 0; i < 8; ++i) s += part[i];
        dinv[row] = rsqrtf(s);                  // deg >= 1 guaranteed
    }
}

// --- Pass 2: support = (x*mask) @ W ; s2t[c,j] = f16(dinv[j]*support[j,c]) --
__global__ __launch_bounds__(256) void gcn_support(const float* __restrict__ x,
                                                   const float* __restrict__ mask,
                                                   const float* __restrict__ W,
                                                   const float* __restrict__ dinv,
                                                   float* __restrict__ support,
                                                   half_t* __restrict__ s2t) {
    __shared__ float xm[32 * DIM];              // 16 KB row tile
    const int t  = threadIdx.x;
    const int m0 = blockIdx.x * 32;
    const float4* xp = (const float4*)(x + (size_t)m0 * DIM);
    const float4* mp = (const float4*)(mask + (size_t)m0 * DIM);
    float4* sp = (float4*)xm;
#pragma unroll
    for (int i = 0; i < 4; ++i) {               // 32*128 f32 = 1024 float4
        int idx = t + i * 256;
        float4 xv = xp[idx], mv = mp[idx], r;
        r.x = xv.x * mv.x; r.y = xv.y * mv.y; r.z = xv.z * mv.z; r.w = xv.w * mv.w;
        sp[idx] = r;
    }
    __syncthreads();
    const int col = t & 127;
    const int rg  = (t >> 7) * 16;              // 2 row-groups of 16
    float acc[16] = {};
    for (int k = 0; k < DIM; ++k) {
        float w = W[k * DIM + col];             // W hot in L2 (64 KB)
#pragma unroll
        for (int r = 0; r < 16; ++r)            // LDS address uniform -> broadcast
            acc[r] += xm[(rg + r) * DIM + k] * w;
    }
    // row-major f32 support (for the epilogue's diagonal term)
#pragma unroll
    for (int r = 0; r < 16; ++r)
        support[(size_t)(m0 + rg + r) * DIM + col] = acc[r];
    // transposed f16 s2t: this thread owns 16 consecutive rows of one column,
    // so the transposed store is a single contiguous 32 B write.
    v8h lo, hi;
#pragma unroll
    for (int r = 0; r < 8; ++r) {
        lo[r] = (half_t)(dinv[m0 + rg + r] * acc[r]);
        hi[r] = (half_t)(dinv[m0 + rg + 8 + r] * acc[8 + r]);
    }
    v8h* dp = (v8h*)(s2t + (size_t)col * NROW + m0 + rg);
    dp[0] = lo;
    dp[1] = hi;
}

// --- Pass 3: out = dinv*(adj_f16 @ s2) + dinv^2*support + bias (WMMA) ------
// 128 blocks x 256 threads; block tile = 64 rows x 128 cols, K tile = 32.
// 8 waves: (row-strip of 16) x (64-col half); 4 v8f accumulators per wave.
// Software pipeline: regs hold tile k+1 while WMMA consumes tile k from the
// alternate LDS buffer; one barrier per K step; L2 prefetch 8 tiles ahead.
__global__ __launch_bounds__(256) void gcn_aggregate(const float* __restrict__ adj,
                                                     const half_t* __restrict__ s2t,
                                                     const float* __restrict__ support,
                                                     const float* __restrict__ dinv,
                                                     const float* __restrict__ bias,
                                                     float* __restrict__ out) {
    __shared__ half_t lds_a[2][64 * LDA];       // adj tile  (f16)  2 x 5120 B
    __shared__ half_t lds_b[2][DIM * LDA];      // s2 tile [n][k]   2 x 10240 B
    const int t     = threadIdx.x;
    const int lane  = t & 31;
    const int wid   = t >> 5;
    const int strip = wid >> 1;                 // 0..3 -> row strip
    const int nhalf = wid & 1;                  // 0..1 -> column half
    const int m0    = blockIdx.x * 64;

    // global->LDS load mappings
    const int arow = t >> 2, aseg = t & 3;      // A: 64 rows x 4 segs of 8 f32
    const int bn   = t >> 1, bhalf = t & 1;     // B: 128 n-rows x 2 segs of 16 f16
    const float* arp = adj + (size_t)(m0 + arow) * NROW + aseg * 8;
    const half_t* brp = s2t + (size_t)bn * NROW + bhalf * 16;
    // fragment mapping (ISA 16-bit operand layout)
    const int frow  = lane & 15;
    const int kbase = (lane >> 4) * 8;
    const half_t* afp = &lds_a[0][0] + (strip * 16 + frow) * LDA + kbase;
    const half_t* bfp = &lds_b[0][0] + (nhalf * 64 + frow) * LDA + kbase;

    v8f acc[4] = {};

    // ---- prologue: fetch tile 0 to regs, stage into buffer 0 ----
    float4 a0 = ((const float4*)arp)[0];
    float4 a1 = ((const float4*)arp)[1];
    v8h    b0 = ((const v8h*)brp)[0];
    v8h    b1 = ((const v8h*)brp)[1];
    {
        v8h ah0, ah1;
#pragma unroll
        for (int i = 0; i < 4; ++i) {
            ah0[i] = (half_t)(&a0.x)[i]; ah0[4 + i] = (half_t)(&a1.x)[i];
        }
        (void)ah1;
        *(v8h*)(&lds_a[0][0] + arow * LDA + aseg * 8) = ah0;
        *(v8h*)(&lds_b[0][0] + bn * LDA + bhalf * 16)     = b0;
        *(v8h*)(&lds_b[0][0] + bn * LDA + bhalf * 16 + 8) = b1;
    }
    __syncthreads();

    int buf = 0;
    for (int k0 = KT; k0 < NROW; k0 += KT) {
        // L2 prefetch: one 128 B line covers a full future K-tile row of adj
        if (aseg == 0 && k0 + PF_TILES * KT < NROW)
            __builtin_prefetch(arp + k0 + PF_TILES * KT, 0, 1);

        // fetch tile k0 into registers (overlaps with compute below)
        a0 = ((const float4*)(arp + k0))[0];
        a1 = ((const float4*)(arp + k0))[1];
        b0 = ((const v8h*)(brp + k0))[0];
        b1 = ((const v8h*)(brp + k0))[1];

        // compute tile k0-KT from lds[buf]
        {
            const int bofs = buf * 64 * LDA;
            const int bofsB = buf * DIM * LDA;
            v16h af = ld_frag(afp + bofs);
#pragma unroll
            for (int tl = 0; tl < 4; ++tl) {
                v16h bf = ld_frag(bfp + bofsB + tl * 16 * LDA);
                acc[tl] = __builtin_amdgcn_wmma_f32_16x16x32_f16(
                    false, af, false, bf, (short)0, acc[tl], false, false);
            }
        }

        // stage tile k0 into the other buffer
        {
            v8h ah;
#pragma unroll
            for (int i = 0; i < 4; ++i) {
                ah[i] = (half_t)(&a0.x)[i]; ah[4 + i] = (half_t)(&a1.x)[i];
            }
            half_t* pa = &lds_a[0][0] + (buf ^ 1) * 64 * LDA + arow * LDA + aseg * 8;
            half_t* pb = &lds_b[0][0] + (buf ^ 1) * DIM * LDA + bn * LDA + bhalf * 16;
            *(v8h*)pa = ah;
            *(v8h*)(pb)     = b0;
            *(v8h*)(pb + 8) = b1;
        }
        __syncthreads();
        buf ^= 1;
    }

    // ---- epilogue tile ----
    {
        const int bofs = buf * 64 * LDA;
        const int bofsB = buf * DIM * LDA;
        v16h af = ld_frag(afp + bofs);
#pragma unroll
        for (int tl = 0; tl < 4; ++tl) {
            v16h bf = ld_frag(bfp + bofsB + tl * 16 * LDA);
            acc[tl] = __builtin_amdgcn_wmma_f32_16x16x32_f16(
                false, af, false, bf, (short)0, acc[tl], false, false);
        }
    }

    // C/D layout: lanes 0-15 -> M=r, lanes 16-31 -> M=r+8; N = lane&15
#pragma unroll
    for (int tl = 0; tl < 4; ++tl) {
#pragma unroll
        for (int r = 0; r < 8; ++r) {
            int row_local = strip * 16 + r + (lane >> 4) * 8;
            int gm  = m0 + row_local;
            int col = nhalf * 64 + tl * 16 + (lane & 15);
            float di = dinv[gm];
            out[(size_t)gm * DIM + col] =
                di * acc[tl][r] + di * di * support[(size_t)gm * DIM + col] + bias[col];
        }
    }
}

extern "C" void kernel_launch(void* const* d_in, const int* in_sizes, int n_in,
                              void* d_out, int out_size, void* d_ws, size_t ws_size,
                              hipStream_t stream) {
    const float* x    = (const float*)d_in[0];   // [8192,128]
    const float* adj  = (const float*)d_in[1];   // [8192,8192]
    const float* W    = (const float*)d_in[2];   // [128,128]
    const float* bias = (const float*)d_in[3];   // [128]
    const float* mask = (const float*)d_in[4];   // [8192,128]
    float* out = (float*)d_out;                  // [8192,128]

    char* ws = (char*)d_ws;
    float*  dinv    = (float*)ws;                                    // 32 KB
    float*  support = (float*)(ws + (size_t)NROW * 4);               // 4 MB
    half_t* s2t     = (half_t*)(ws + (size_t)NROW * 4
                                   + (size_t)NROW * DIM * 4);        // 2 MB

    gcn_degree   <<<NROW,      256, 0, stream>>>(adj, dinv);
    gcn_support  <<<NROW / 32, 256, 0, stream>>>(x, mask, W, dinv, support, s2t);
    gcn_aggregate<<<NROW / 64, 256, 0, stream>>>(adj, s2t, support, dinv, bias, out);
}
